// TransformerVAEDecoder_43903155700409
// MI455X (gfx1250) — matrix operand order repaired and checked by reading, not compiled, not run on hardware
//
#include <hip/hip_runtime.h>
#include <hip/hip_bf16.h>

// ---------------------------------------------------------------------------
// GRU decoder, persistent wave-cluster kernel for gfx1250 (MI455X).
//   B=64, E=512, 3E=1536, S=2048, latent=128.
// Recurrent GEMM h(64x512) @ W_hh^T(512x1536) runs in bf16 WMMA (f32 accum),
// W_hh resident in LDS (96KB/WGP across 16 WGPs), h exchanged via L2,
// one cluster barrier (+ atomic fallback) per step.
// ---------------------------------------------------------------------------

typedef __attribute__((ext_vector_type(16))) __bf16 bf16x16;
typedef __attribute__((ext_vector_type(8)))  float  f32x8;

#define LATENT 128
#define EMB    512
#define G3     1536
#define BATCH  64
#define SEQ    2048
#define NWG    16          // workgroups == cluster size
#define TPB    256         // 8 waves of 32
#define HSL    32          // hidden units per WG
#define NSL    96          // gate columns per WG (3*HSL)
#define KT     16          // K tiles (512/32)
#define NT     6           // N tiles (96/16)

// ---- workspace layout (bytes) ----
#define OFF_X      ((size_t)0)                       // 64*512 f32      = 131072
#define OFF_GI     ((size_t)131072)                  // 64*1536 f32     = 393216
#define OFF_WPACK  ((size_t)524288)                  // 1536*512 bf16   = 1572864
#define OFF_H0     ((size_t)2097152)                 // 64*512 bf16     = 65536
#define OFF_H1     ((size_t)2162688)                 // 64*512 bf16     = 65536
#define OFF_BAR    ((size_t)2228224)                 // 2 u32 (+pad)
#define OFF_PART   ((size_t)2228480)                 // 2048*16*64 f32  = 8388608

// LDS partition (dynamic): gh 24576 | gi_s 24576 | h_old 8192 | bhh 384 |
//                          wout 128 | red 1024 | wB 98304  => 157184 bytes
#define SMEM_BYTES 157184

__device__ __forceinline__ unsigned short f2bf(float f) {
  unsigned u = __float_as_uint(f);
  unsigned r = (u + 0x7FFFu + ((u >> 16) & 1u)) >> 16;   // RNE
  return (unsigned short)r;
}

union AFrag { bf16x16 v; int4 q[2]; };
union BFrag { bf16x16 v; int4 q[2]; };

__device__ __forceinline__ void cluster_grid_barrier(unsigned* bar) {
  __threadfence();                 // release our global h/partial writes (agent)
  __syncthreads();
  if (threadIdx.x == 0) {          // software inter-WGP barrier (always correct)
    unsigned g = __hip_atomic_load(bar + 1, __ATOMIC_RELAXED, __HIP_MEMORY_SCOPE_AGENT);
    unsigned a = __hip_atomic_fetch_add(bar, 1u, __ATOMIC_ACQ_REL, __HIP_MEMORY_SCOPE_AGENT);
    if (a == NWG - 1) {
      __hip_atomic_store(bar,     0u,     __ATOMIC_RELAXED, __HIP_MEMORY_SCOPE_AGENT);
      __hip_atomic_store(bar + 1, g + 1u, __ATOMIC_RELEASE, __HIP_MEMORY_SCOPE_AGENT);
    } else {
      while (__hip_atomic_load(bar + 1, __ATOMIC_ACQUIRE, __HIP_MEMORY_SCOPE_AGENT) == g)
        __builtin_amdgcn_s_sleep(1);
    }
  }
  // HW fast path when dispatched as a workgroup-cluster (s_barrier_signal/wait -3;
  // NOP when ClusterID==0).
  __builtin_amdgcn_s_cluster_barrier();
  __syncthreads();
  __builtin_amdgcn_fence(__ATOMIC_ACQUIRE, "agent");   // invalidate stale WGP$ lines
}

// ---------------------------------------------------------------------------
// Prep kernels
// ---------------------------------------------------------------------------
__global__ void k_emb(const float* __restrict__ z, const float* __restrict__ W_emb,
                      const float* __restrict__ b_emb, float* __restrict__ x) {
  int t = blockIdx.x * blockDim.x + threadIdx.x;
  if (t >= BATCH * EMB) return;
  int b = t / EMB, e = t % EMB;
  float a = b_emb[e];
  const float* zr = z + b * LATENT;
  const float* wr = W_emb + e * LATENT;
  for (int l = 0; l < LATENT; ++l) a += zr[l] * wr[l];
  x[t] = a;
}

__global__ void k_gi(const float* __restrict__ x, const float* __restrict__ W_ih,
                     const float* __restrict__ b_ih, float* __restrict__ gi) {
  int t = blockIdx.x * blockDim.x + threadIdx.x;
  if (t >= BATCH * G3) return;
  int b = t / G3, n = t % G3;
  float a = b_ih[n];
  const float* xr = x + b * EMB;
  const float* wr = W_ih + n * EMB;
  for (int k = 0; k < EMB; ++k) a += xr[k] * wr[k];
  gi[t] = a;
}

// Pack W_hh (1536x512 f32) -> bf16 WMMA-B fragments, per-WG contiguous.
// Fragment f = kt*NT+nt; lane l dword d holds B[k,n],B[k+1,n] with
// n = nt*16 + (l&15) (local gate-col), k = kt*32 + (l>>4)*16 + 2d.
__global__ void k_pack(const float* __restrict__ W_hh, unsigned* __restrict__ wpack) {
  int t = blockIdx.x * blockDim.x + threadIdx.x;
  if (t >= NWG * KT * NT * 256) return;
  int wg = t / (KT * NT * 256);
  int iw = t % (KT * NT * 256);
  int f  = iw >> 8;
  int nt = f % NT, kt = f / NT;
  int rr = iw & 255;
  int lane = rr >> 3, d = rr & 7;
  int nl   = nt * 16 + (lane & 15);        // 0..95
  int gate = nl >> 5, u = nl & 31;
  int ng   = gate * EMB + wg * HSL + u;    // global W_hh row
  int k    = kt * 32 + (lane >> 4) * 16 + d * 2;
  unsigned lo = f2bf(W_hh[ng * EMB + k]);
  unsigned hi = f2bf(W_hh[ng * EMB + k + 1]);
  wpack[t] = lo | (hi << 16);
}

__global__ void k_init(unsigned short* __restrict__ h0, unsigned short* __restrict__ h1,
                       unsigned* __restrict__ bar) {
  int t = blockIdx.x * blockDim.x + threadIdx.x;
  if (t < BATCH * EMB) { h0[t] = 0; h1[t] = 0; }
  if (t < 2) bar[t] = 0u;
}

__global__ void k_out(const float* __restrict__ partial, const float* __restrict__ b_out,
                      float* __restrict__ out) {
  int t = blockIdx.x * blockDim.x + threadIdx.x;
  if (t >= BATCH * SEQ) return;
  int b = t / SEQ, s = t % SEQ;
  float a = b_out[0];
  const float* p = partial + (size_t)s * NWG * BATCH + b;
#pragma unroll
  for (int w = 0; w < NWG; ++w) a += p[w * BATCH];
  out[t] = a;
}

// ---------------------------------------------------------------------------
// Persistent recurrent kernel: 16 WGs, one per WGP, 2048 steps.
// ---------------------------------------------------------------------------
__global__ void __launch_bounds__(TPB)
k_gru(const float* __restrict__ gi,       // (64,1536) fp32, includes b_ih
      const unsigned* __restrict__ wpack, // packed bf16 B-fragments
      const float* __restrict__ b_hh,     // (1536)
      const float* __restrict__ W_out,    // (512)
      unsigned short* __restrict__ h0,    // bf16 (64,512) double buffer
      unsigned short* __restrict__ h1,
      float* __restrict__ partial,        // (SEQ, NWG, 64)
      unsigned* __restrict__ bar) {
  extern __shared__ char smem[];
  float* gh     = (float*)(smem);                  // 64*96
  float* gi_s   = (float*)(smem + 24576);          // 64*96
  float* h_old  = (float*)(smem + 49152);          // 64*32 fp32 state slice
  float* bhh_s  = (float*)(smem + 57344);          // 96
  float* wout_s = (float*)(smem + 57728);          // 32
  float* red    = (float*)(smem + 57856);          // 256
  unsigned* wB  = (unsigned*)(smem + 58880);       // 96KB of B fragments

  const int wg   = blockIdx.x;
  const int tid  = threadIdx.x;
  const int wave = tid >> 5;
  const int lane = tid & 31;

  // ---- one-time LDS fill: weights, gi slice, biases, W_out slice ----
  {
    const int4* src = (const int4*)(wpack + (size_t)wg * (KT * NT * 256));
    int4* dst = (int4*)wB;
    for (int i = tid; i < KT * NT * 64; i += TPB) dst[i] = src[i];
    for (int i = tid; i < BATCH * NSL; i += TPB) {
      int b = i / NSL, nl = i % NSL;
      int gate = nl >> 5, u = nl & 31;
      gi_s[i] = gi[b * G3 + gate * EMB + wg * HSL + u];
    }
    for (int i = tid; i < NSL; i += TPB) {
      int gate = i >> 5, u = i & 31;
      bhh_s[i] = b_hh[gate * EMB + wg * HSL + u];
    }
    for (int i = tid; i < HSL; i += TPB) wout_s[i] = W_out[wg * HSL + i];
    for (int i = tid; i < BATCH * HSL; i += TPB) h_old[i] = 0.0f;
  }
  __syncthreads();

  const int mtile = wave >> 1;        // 0..3  (16 batch rows each)
  const int nbase = (wave & 1) * 3;   // 3 of 6 N-tiles per wave

#pragma unroll 1
  for (int s = 0; s < SEQ; ++s) {
    const unsigned short* hc = (s & 1) ? h1 : h0;
    unsigned short*       hn = (s & 1) ? h0 : h1;

    // ---- load all A fragments for this wave's M-tile (row-major bf16 h) ----
    AFrag af[KT];
    {
      const int m  = mtile * 16 + (lane & 15);
      const int kh = (lane >> 4) * 16;
      const unsigned short* arow = hc + m * EMB + kh;
#pragma unroll
      for (int kt = 0; kt < KT; ++kt) {
        af[kt].q[0] = *(const int4*)(arow + kt * 32);
        af[kt].q[1] = *(const int4*)(arow + kt * 32 + 8);
      }
    }

    // ---- gh = h @ W_hh^T (slice), bf16 WMMA, fp32 accum ----
    f32x8 acc[3];
#pragma unroll
    for (int j = 0; j < 3; ++j)
#pragma unroll
      for (int r = 0; r < 8; ++r) acc[j][r] = 0.0f;

#pragma unroll
    for (int kt = 0; kt < KT; ++kt) {
#pragma unroll
      for (int j = 0; j < 3; ++j) {
        BFrag bf;
        const int4* bp =
            (const int4*)((const char*)wB + (size_t)(kt * NT + nbase + j) * 1024 + lane * 32);
        bf.q[0] = bp[0];
        bf.q[1] = bp[1];
        acc[j] = __builtin_amdgcn_wmma_f32_16x16x32_bf16(
            false, af[kt].v, false, bf.v, (short)0, acc[j], false, false);
      }
    }

    // ---- spill C tiles to LDS gh[m][ncol] ----
#pragma unroll
    for (int j = 0; j < 3; ++j) {
      const int ncol = (nbase + j) * 16 + (lane & 15);
      const int mrow = mtile * 16 + (lane >> 4) * 8;
#pragma unroll
      for (int r = 0; r < 8; ++r) gh[(mrow + r) * NSL + ncol] = acc[j][r];
    }
    __syncthreads();

    // ---- GRU pointwise update on this WG's 64x32 slice (fp32) ----
    {
      const int b  = tid & 63;
      const int up = tid >> 6;   // 0..3
      float pacc = 0.0f;
#pragma unroll
      for (int jj = 0; jj < 8; ++jj) {
        const int u = up * 8 + jj;
        const float gr  = gi_s[b * NSL + u]      + gh[b * NSL + u]      + bhh_s[u];
        const float gz  = gi_s[b * NSL + 32 + u] + gh[b * NSL + 32 + u] + bhh_s[32 + u];
        const float gni = gi_s[b * NSL + 64 + u];
        const float gnh = gh[b * NSL + 64 + u] + bhh_s[64 + u];
        const float rg = 1.0f / (1.0f + __expf(-gr));
        const float zg = 1.0f / (1.0f + __expf(-gz));
        const float ng = tanhf(gni + rg * gnh);
        const float h  = h_old[b * HSL + u];
        const float hv = (1.0f - zg) * ng + zg * h;
        h_old[b * HSL + u] = hv;
        hn[b * EMB + wg * HSL + u] = f2bf(hv);
        pacc += wout_s[u] * hv;
      }
      red[up * 64 + b] = pacc;
    }
    __syncthreads();
    if (tid < BATCH)
      partial[((size_t)s * NWG + wg) * BATCH + tid] =
          red[tid] + red[64 + tid] + red[128 + tid] + red[192 + tid];

    // ---- one inter-WGP sync per step ----
    cluster_grid_barrier(bar);
  }
}

// ---------------------------------------------------------------------------
extern "C" void kernel_launch(void* const* d_in, const int* in_sizes, int n_in,
                              void* d_out, int out_size, void* d_ws, size_t ws_size,
                              hipStream_t stream) {
  const float* z     = (const float*)d_in[0];
  const float* W_emb = (const float*)d_in[1];
  const float* b_emb = (const float*)d_in[2];
  const float* W_ih  = (const float*)d_in[3];
  const float* b_ih  = (const float*)d_in[4];
  const float* W_hh  = (const float*)d_in[5];
  const float* b_hh  = (const float*)d_in[6];
  const float* W_out = (const float*)d_in[7];
  const float* b_out = (const float*)d_in[8];
  float* out = (float*)d_out;

  char* ws = (char*)d_ws;
  float*          x       = (float*)(ws + OFF_X);
  float*          gi      = (float*)(ws + OFF_GI);
  unsigned*       wpack   = (unsigned*)(ws + OFF_WPACK);
  unsigned short* h0      = (unsigned short*)(ws + OFF_H0);
  unsigned short* h1      = (unsigned short*)(ws + OFF_H1);
  unsigned*       bar     = (unsigned*)(ws + OFF_BAR);
  float*          partial = (float*)(ws + OFF_PART);

  k_emb<<<(BATCH * EMB + TPB - 1) / TPB, TPB, 0, stream>>>(z, W_emb, b_emb, x);
  k_gi<<<(BATCH * G3 + TPB - 1) / TPB, TPB, 0, stream>>>(x, W_ih, b_ih, gi);
  k_pack<<<(NWG * KT * NT * 256 + TPB - 1) / TPB, TPB, 0, stream>>>(W_hh, wpack);
  k_init<<<(BATCH * EMB + TPB - 1) / TPB, TPB, 0, stream>>>(h0, h1, bar);
  k_gru<<<NWG, TPB, SMEM_BYTES, stream>>>(gi, wpack, b_hh, W_out, h0, h1, partial, bar);
  k_out<<<(BATCH * SEQ + TPB - 1) / TPB, TPB, 0, stream>>>(partial, b_out, out);
}
